// GCNNet_83502754169548
// MI455X (gfx1250) — matrix operand order repaired and checked by reading, not compiled, not run on hardware
//
#include <hip/hip_runtime.h>

// ---------------- types for WMMA fragments ----------------
typedef __attribute__((ext_vector_type(16))) __bf16 v16bf;
typedef __attribute__((ext_vector_type(8)))  float  v8f;

struct B32x8 { uint4 lo, hi; };   // 32 bytes, bit-cast to v16bf

static __device__ __forceinline__ unsigned short f32_to_bf16(float f) {
    unsigned u = __float_as_uint(f);
    unsigned rnd = 0x7fffu + ((u >> 16) & 1u);   // round-to-nearest-even
    return (unsigned short)((u + rnd) >> 16);
}

// ---------------- elementwise cast / pad / relu ----------------
__global__ void cast_pad_bf16(const float* __restrict__ in, unsigned short* __restrict__ out,
                              int rin, int cin, int rout, int cout, int relu) {
    size_t i = (size_t)blockIdx.x * blockDim.x + threadIdx.x;
    size_t tot = (size_t)rout * cout;
    if (i >= tot) return;
    int r = (int)(i / cout);
    int c = (int)(i - (size_t)r * cout);
    float v = 0.0f;
    if (r < rin && c < cin) v = in[(size_t)r * cin + c];
    if (relu) v = fmaxf(v, 0.0f);
    out[i] = f32_to_bf16(v);
}

// cast weights W[K x N] (f32, row-major) -> WT[NP x KP] (bf16, transposed, zero-padded)
__global__ void cast_pad_bf16_T(const float* __restrict__ in, unsigned short* __restrict__ out,
                                int kin, int nin, int nout, int kout) {
    size_t i = (size_t)blockIdx.x * blockDim.x + threadIdx.x;
    size_t tot = (size_t)nout * kout;
    if (i >= tot) return;
    int n = (int)(i / kout);
    int k = (int)(i - (size_t)n * kout);
    float v = 0.0f;
    if (k < kin && n < nin) v = in[(size_t)k * nin + n];
    out[i] = f32_to_bf16(v);
}

// ---------------- degree / normalization ----------------
__global__ void deg_init(float* __restrict__ deg, int n) {
    int i = blockIdx.x * blockDim.x + threadIdx.x;
    if (i < n) deg[i] = 1.0f;                       // self-loop contribution
}
__global__ void deg_accum(const int* __restrict__ dst, float* __restrict__ deg, int E) {
    int e = blockIdx.x * blockDim.x + threadIdx.x;
    if (e < E) atomicAdd(&deg[dst[e]], 1.0f);
}
__global__ void deg_rsqrt(float* __restrict__ deg, int n) {
    int i = blockIdx.x * blockDim.x + threadIdx.x;
    if (i < n) deg[i] = rsqrtf(deg[i]);             // deg >= 1 always
}

// ---------------- WMMA bf16 GEMM: C[M x N] = A[M x K] * BT[N x K]^T ----------------
// Block tile 128x64, 8 waves, each wave 32x32 (2x2 WMMA tiles), K step 32.
// BT is the pre-transposed weight matrix: BT[n][k] = B[k][n], row stride K.
#define LDA_S 40   // LDS row stride in bf16 elems (80B = 20 banks, conflict-free)
#define LDB_S 40

__global__ __launch_bounds__(256)
void wmma_gemm_bf16(const unsigned short* __restrict__ A,
                    const unsigned short* __restrict__ BT,
                    float* __restrict__ C, int N, int K) {
    __shared__ __align__(16) unsigned short lA[128 * LDA_S];
    __shared__ __align__(16) unsigned short lB[64 * LDB_S];   // [n][k]

    const int tid  = threadIdx.x;
    const int wid  = tid >> 5;
    const int lane = tid & 31;
    const int lr   = lane & 15;        // fragment row/col
    const int lh   = lane >> 4;        // lane half
    const int bm   = blockIdx.x * 128;
    const int bn   = blockIdx.y * 64;
    const int wm   = (wid & 3) * 32;   // wave offset in tile (M)
    const int wn   = (wid >> 2) * 32;  // wave offset in tile (N)

    v8f c00 = {}, c01 = {}, c10 = {}, c11 = {};

    for (int k0 = 0; k0 < K; k0 += 32) {
        // ---- stage A tile 128x32 (row-major), 2 threads per row, 32B each ----
        {
            int r = tid >> 1;
            int s = (tid & 1) * 16;                       // bf16 offset in row
            const unsigned short* g = A + (size_t)(bm + r) * K + k0 + s;
            uint4 d0 = *(const uint4*)g;
            uint4 d1 = *(const uint4*)(g + 8);
            *(uint4*)&lA[r * LDA_S + s]     = d0;
            *(uint4*)&lA[r * LDA_S + s + 8] = d1;
        }
        // ---- stage B tile 64(n) x 32(k) from pre-transposed BT: coalesced b128 ----
        {
            int n = tid >> 2;                             // 0..63
            int s = (tid & 3) * 8;                        // 0..24 (bf16 elems)
            const unsigned short* g = BT + (size_t)(bn + n) * K + k0 + s;
            uint4 d = *(const uint4*)g;
            *(uint4*)&lB[n * LDB_S + s] = d;
        }
        __syncthreads();

        // ---- load fragments (CDNA5 16-bit VGPR layouts, CDNA5 ISA §7.12.2) ----
        B32x8 t;
        const unsigned short* p;

        p = &lA[(wm + lr) * LDA_S + lh * 8];              // A rows wm..wm+15
        t.lo = *(const uint4*)p;  t.hi = *(const uint4*)(p + 16);
        v16bf a0 = __builtin_bit_cast(v16bf, t);

        p = &lA[(wm + 16 + lr) * LDA_S + lh * 8];         // A rows wm+16..wm+31
        t.lo = *(const uint4*)p;  t.hi = *(const uint4*)(p + 16);
        v16bf a1 = __builtin_bit_cast(v16bf, t);

        p = &lB[(wn + lr) * LDB_S + lh * 8];              // B cols wn..wn+15
        t.lo = *(const uint4*)p;  t.hi = *(const uint4*)(p + 16);
        v16bf b0 = __builtin_bit_cast(v16bf, t);

        p = &lB[(wn + 16 + lr) * LDB_S + lh * 8];         // B cols wn+16..wn+31
        t.lo = *(const uint4*)p;  t.hi = *(const uint4*)(p + 16);
        v16bf b1 = __builtin_bit_cast(v16bf, t);

        c00 = __builtin_amdgcn_wmma_f32_16x16x32_bf16(false, a0, false, b0, (short)0, c00, false, false);
        c01 = __builtin_amdgcn_wmma_f32_16x16x32_bf16(false, a0, false, b1, (short)0, c01, false, false);
        c10 = __builtin_amdgcn_wmma_f32_16x16x32_bf16(false, a1, false, b0, (short)0, c10, false, false);
        c11 = __builtin_amdgcn_wmma_f32_16x16x32_bf16(false, a1, false, b1, (short)0, c11, false, false);
        __syncthreads();
    }

    // ---- epilogue: f32 C/D layout — VGPR r: lanes0-15 row r, lanes16-31 row r+8 ----
#pragma unroll
    for (int r = 0; r < 8; ++r) {
        int row0 = bm + wm + r + 8 * lh;
        int row1 = row0 + 16;
        int col0 = bn + wn + lr;
        C[(size_t)row0 * N + col0]      = c00[r];
        C[(size_t)row0 * N + col0 + 16] = c01[r];
        C[(size_t)row1 * N + col0]      = c10[r];
        C[(size_t)row1 * N + col0 + 16] = c11[r];
    }
}

// ---------------- aggregation: out = bias + selfloop + sum over edges ----------------
__global__ void agg_init(const float* __restrict__ h, int hstride,
                         const float* __restrict__ dinv, const float* __restrict__ bias,
                         float* __restrict__ out, int n, int F) {
    unsigned idx = blockIdx.x * blockDim.x + threadIdx.x;
    unsigned tot = (unsigned)n * (unsigned)F;
    if (idx >= tot) return;
    int r = idx / F;
    int f = idx - r * F;
    float di = dinv[r];
    out[idx] = bias[f] + h[(size_t)r * hstride + f] * di * di;   // self-loop norm = dinv^2
}

__global__ void agg_edges(const int* __restrict__ src, const int* __restrict__ dst,
                          const float* __restrict__ dinv,
                          const float* __restrict__ h, int hstride,
                          float* __restrict__ out, int F, int E) {
    unsigned idx = blockIdx.x * blockDim.x + threadIdx.x;
    unsigned tot = (unsigned)E * (unsigned)F;
    if (idx >= tot) return;
    int e = idx / F;
    int f = idx - e * F;
    int s = src[e];
    int d = dst[e];
    float w = dinv[s] * dinv[d];
    atomicAdd(out + (size_t)d * F + f, h[(size_t)s * hstride + f] * w);
}

// ---------------- launch ----------------
extern "C" void kernel_launch(void* const* d_in, const int* in_sizes, int n_in,
                              void* d_out, int out_size, void* d_ws, size_t ws_size,
                              hipStream_t stream) {
    const float* x  = (const float*)d_in[0];
    const int*   ei = (const int*)d_in[1];
    const float* W1 = (const float*)d_in[2];
    const float* b1 = (const float*)d_in[3];
    const float* W2 = (const float*)d_in[4];
    const float* b2 = (const float*)d_in[5];
    float* out = (float*)d_out;

    const int HID = in_sizes[3];            // 512
    const int FIN = in_sizes[2] / HID;      // 512
    const int NN  = in_sizes[0] / FIN;      // 20000
    const int E   = in_sizes[1] / 2;        // 160000
    const int NC  = in_sizes[5];            // 40
    const int MP  = ((NN + 127) / 128) * 128;   // 20096 (WMMA tile padding)
    const int NCP = 64;                     // N_CLASSES padded to tile width

    const int* srcE = ei;
    const int* dstE = ei + E;

    char* ws = (char*)d_ws;
    size_t off = 0;
    auto alloc = [&](size_t bytes) -> void* {
        void* p = ws + off;
        off += (bytes + 255) & ~(size_t)255;
        return p;
    };
    unsigned short* xb   = (unsigned short*)alloc((size_t)MP * FIN * 2);  // reused as relu-bf16 later
    unsigned short* w1t  = (unsigned short*)alloc((size_t)HID * FIN * 2); // W1^T [HID][FIN]
    unsigned short* w2t  = (unsigned short*)alloc((size_t)NCP * HID * 2); // W2^T [NCP][HID]
    float*          dinv = (float*)alloc((size_t)NN * 4);
    float*          h1   = (float*)alloc((size_t)MP * HID * 4);           // reused as h2 later
    float*          agg1 = (float*)alloc((size_t)NN * HID * 4);
    (void)ws_size; (void)n_in; (void)out_size;

    auto blocks = [](size_t tot) { return (unsigned)((tot + 255) / 256); };

    // casts to bf16 (x padded; weights transposed + padded)
    cast_pad_bf16  <<<blocks((size_t)MP * FIN),  256, 0, stream>>>(x,  xb,  NN,  FIN, MP,  FIN, 0);
    cast_pad_bf16_T<<<blocks((size_t)HID * FIN), 256, 0, stream>>>(W1, w1t, FIN, HID, HID, FIN);
    cast_pad_bf16_T<<<blocks((size_t)NCP * HID), 256, 0, stream>>>(W2, w2t, HID, NC,  NCP, HID);

    // degree -> dinv
    deg_init <<<blocks(NN), 256, 0, stream>>>(dinv, NN);
    deg_accum<<<blocks(E),  256, 0, stream>>>(dstE, dinv, E);
    deg_rsqrt<<<blocks(NN), 256, 0, stream>>>(dinv, NN);

    // layer 1: h1 = x @ W1 (WMMA), then normalized aggregation + bias
    wmma_gemm_bf16<<<dim3(MP / 128, HID / 64), 256, 0, stream>>>(xb, w1t, h1, HID, FIN);
    agg_init <<<blocks((size_t)NN * HID), 256, 0, stream>>>(h1, HID, dinv, b1, agg1, NN, HID);
    agg_edges<<<blocks((size_t)E  * HID), 256, 0, stream>>>(srcE, dstE, dinv, h1, HID, agg1, HID, E);

    // relu + cast (reuse xb as bf16 hidden activations)
    cast_pad_bf16<<<blocks((size_t)MP * HID), 256, 0, stream>>>(agg1, xb, NN, HID, MP, HID, 1);

    // layer 2: h2 = relu(h) @ W2 (WMMA, N padded to 64; reuse h1 buffer)
    wmma_gemm_bf16<<<dim3(MP / 128, NCP / 64), 256, 0, stream>>>(xb, w2t, h1, NCP, HID);
    agg_init <<<blocks((size_t)NN * NC), 256, 0, stream>>>(h1, NCP, dinv, b2, out, NN, NC);
    agg_edges<<<blocks((size_t)E  * NC), 256, 0, stream>>>(srcE, dstE, dinv, h1, NCP, out, NC, E);
}
